// GCNTCN_2018634629423
// MI455X (gfx1250) — matrix operand order repaired and checked by reading, not compiled, not run on hardware
//
#include <hip/hip_runtime.h>

// ---------------------------------------------------------------------------
// GCN-TCN forward for gfx1250 (MI455X).
// Dense contractions -> v_wmma_f32_16x16x32_f16 with LDS-staged tiles.
// GEMM weight panels are pre-transposed once so per-chunk staging is a pure
// contiguous 2D tile copy -> fed through the Tensor Data Mover
// (tensor_load_to_lds + s_wait_tensorcnt) when the builtin is available.
//
// Input index map (setup_inputs dict order, nested insertion order):
//  0 data[256,64,512] f32 | 1 ei0[2,1024] i32 | 2 ei1[2,1024] i32
//  TCN blk1: 3 v1 4 g1 5 b1 6 v2 7 g2 8 b2 9 dw 10 db
//  TCN blk2: 11 v1 12 g1 13 b1 14 v2 15 g2 16 b2
//  TCN blk3: 17 v1 18 g1 19 b1 20 v2 21 g2 22 b2 23 dw 24 db
//  GCN0: 25 w1 26 b1 27 w2 28 b2 29 ln1g 30 ln1b 31 ln2g 32 ln2b 33 pw 34 pb
//  GCN1: 35..44 | head: 45 w1 46 b1 47 w2 48 b2
// ---------------------------------------------------------------------------

typedef __attribute__((ext_vector_type(16))) _Float16 v16h;
typedef __attribute__((ext_vector_type(8)))  _Float16 v8h;
typedef __attribute__((ext_vector_type(4)))  _Float16 v4h;
typedef __attribute__((ext_vector_type(8)))  float    v8f;
typedef __attribute__((ext_vector_type(4)))  float    v4f;
typedef __attribute__((ext_vector_type(4)))  unsigned int v4u;
typedef __attribute__((ext_vector_type(8)))  int      v8i;
typedef __attribute__((ext_vector_type(4)))  int      v4i;

#define BB   256
#define NN   64
#define TT   512
#define HIDc 48
#define OUTc 256
#define EE   1024

#if __has_builtin(__builtin_amdgcn_tensor_load_to_lds) && __has_builtin(__builtin_amdgcn_s_wait_tensorcnt)
#define HAVE_TDM 1
#else
#define HAVE_TDM 0
#endif

// ---------------------------------------------------------------------------
// Weight-norm prep: packs weight_norm'd conv weights as f16 [tap][cout][cin]
__global__ void wn_prep(const float* __restrict__ v, const float* __restrict__ g,
                        _Float16* __restrict__ out, int cout, int cin, int ntaps) {
    int m = blockIdx.x * blockDim.x + threadIdx.x;
    if (m >= cout) return;
    const float* vm = v + (size_t)m * cin * ntaps;
    float ss = 0.f;
    for (int i = 0; i < cin * ntaps; ++i) ss += vm[i] * vm[i];
    float s = g[m] * rsqrtf(ss);
    for (int ci = 0; ci < cin; ++ci)
        for (int k = 0; k < ntaps; ++k)
            out[((size_t)k * cout + m) * cin + ci] = (_Float16)(vm[ci * ntaps + k] * s);
}

__global__ void cvt_f16(const float* __restrict__ in, _Float16* __restrict__ out, int n) {
    int i = blockIdx.x * blockDim.x + threadIdx.x;
    if (i < n) out[i] = (_Float16)in[i];
}

// transpose-convert: in[K][N] f32 -> out[N][K] f16 (done once; makes GEMM
// B-tiles contiguous 2D strided copies == TDM-friendly)
__global__ void cvt_f16_t(const float* __restrict__ in, _Float16* __restrict__ out,
                          int K, int N) {
    int idx = blockIdx.x * blockDim.x + threadIdx.x;
    if (idx >= K * N) return;
    int k = idx / N, n = idx - k * N;
    out[(size_t)n * K + k] = (_Float16)in[idx];
}

// ---------------------------------------------------------------------------
// Dilated causal conv as WMMA GEMM.  Block = COUT/16 waves; f16 activation
// tile [16+PAD cols x CIN] staged ONCE in LDS (t-major, zero-padded left
// edge), shared by all waves/taps/chunks.  A-fragments = contiguous b128
// weight loads; B-fragments = one aligned 32B LDS read per lane.
template<int CIN, int COUT, int NTAPS, int DIL, int RELU, int ADDRES>
__global__ __launch_bounds__((COUT/16)*32)
void conv_wmma_t(const float* __restrict__ X, const _Float16* __restrict__ W,
                 const float* __restrict__ bias, const float* __restrict__ res,
                 float* __restrict__ Y) {
    constexpr int PAD = (NTAPS - 1) * DIL;
    constexpr int TW  = 16 + PAD;
    constexpr int NT  = (COUT / 16) * 32;
    __shared__ __align__(32) _Float16 ldsX[TW * CIN];

    const int tid   = threadIdx.x;
    const int b     = blockIdx.y;
    const int tileT = blockIdx.x << 4;
    const float* Xb = X + (size_t)b * CIN * TT;

    for (int idx = tid; idx < CIN * TW; idx += NT) {
        int ci = idx / TW, tl = idx - ci * TW;
        int gcol = tileT - PAD + tl;
        float v = (gcol >= 0) ? Xb[(size_t)ci * TT + gcol] : 0.f;
        ldsX[tl * CIN + ci] = (_Float16)v;
    }
    __syncthreads();

    const int w = tid >> 5, lane = tid & 31;
    const int h = lane >> 4, r = lane & 15;
    v8f acc = {};
    #pragma unroll
    for (int k = 0; k < NTAPS; ++k) {
        const int shift = (NTAPS - 1 - k) * DIL;
        const int tl = PAD + r - shift;
        #pragma unroll
        for (int kk = 0; kk < CIN; kk += 32) {
            const _Float16* wp = W + ((size_t)k * COUT + 16 * w + r) * CIN + kk + 16 * h;
            v8h alo = *(const v8h*)wp;
            v8h ahi = *(const v8h*)(wp + 8);
            v16h a = __builtin_shufflevector(alo, ahi, 0,1,2,3,4,5,6,7,8,9,10,11,12,13,14,15);
            v16h bf = *(const v16h*)&ldsX[tl * CIN + kk + 16 * h];
            acc = __builtin_amdgcn_wmma_f32_16x16x32_f16(false, a, false, bf,
                                                         (short)0, acc, false, false);
        }
    }

    float* Yb = Y + (size_t)b * COUT * TT;
    const float* Rb = res + (size_t)b * COUT * TT;
    #pragma unroll
    for (int i = 0; i < 8; ++i) {
        int m = 16 * w + i + 8 * h;
        int t = tileT + r;
        float vv = acc[i] + bias[m];
        if (ADDRES) vv += Rb[(size_t)m * TT + t];
        if (RELU) vv = fmaxf(vv, 0.f);
        Yb[(size_t)m * TT + t] = vv;
    }
}

// ---------------------------------------------------------------------------
// Row-major GEMM: C[M,N] = A[M,K](f32) @ W(f16, pre-transposed [N][K]).
// Block = N/16 waves for one 16-row M tile.  Per 32-K chunk: A converted once
// to f16 in LDS (vector loads); the [N x 32] weight panel is DMA'd into LDS
// by the Tensor Data Mover (fallback: vectorized b128 copies).
template<int K, int N, int RELU, int BIAS>
__global__ __launch_bounds__((N/16)*32)
void gemm_wmma_t(const float* __restrict__ A, const _Float16* __restrict__ Wt,
                 const float* __restrict__ bias, float* __restrict__ C) {
    constexpr int NT = (N / 16) * 32;
    __shared__ __align__(32) _Float16 ldsA[16 * 32];
    __shared__ __align__(32) _Float16 ldsB[N * 32];   // [n][k-chunk]

    const int tid = threadIdx.x;
    const int tileM = blockIdx.x << 4;
    const int w = tid >> 5, lane = tid & 31;
    const int h = lane >> 4, r = lane & 15;
    v8f acc = {};

    for (int kk = 0; kk < K; kk += 32) {
        __syncthreads();
        // ---- stage A chunk 16x32 (f32 -> f16), vector loads, zero K-tail ---
        for (int idx = tid; idx < 16 * 8; idx += NT) {
            int m = idx >> 3, c4 = (idx & 7) * 4;
            int k0 = kk + c4;
            const float* sp = A + (size_t)(tileM + m) * K + k0;
            v4h d;
            if (k0 + 3 < K) {
                __builtin_prefetch(sp + 32, 0, 0);          // global_prefetch_b8
                v4f vv = *(const v4f*)sp;
                #pragma unroll
                for (int j = 0; j < 4; ++j) d[j] = (_Float16)vv[j];
            } else {
                #pragma unroll
                for (int j = 0; j < 4; ++j) d[j] = (_Float16)((k0 + j < K) ? sp[j] : 0.f);
            }
            *(v4h*)&ldsA[m * 32 + c4] = d;
        }
        // ---- stage B panel [N x 32] ---------------------------------------
#if HAVE_TDM
        if (tid < 32) {
            // Tensor DMA descriptor (CDNA5 ISA):
            // group0: count=1 | lds_addr | global_addr | type=2
            unsigned long long ga = (unsigned long long)(size_t)Wt + (unsigned)(kk * 2);
            unsigned lds_off = (unsigned)(size_t)(void*)ldsB;
            v4u g0 = { 1u, lds_off, (unsigned)ga,
                       (unsigned)(ga >> 32) | (2u << 30) };
            // group1: data_size=2B; tensor_dim0=K-kk, dim1=N; tile 32 x N;
            // tensor_dim0_stride = K elements
            unsigned td0 = (unsigned)(K - kk);
            v8i g1 = { (int)(1u << 16),
                       (int)((td0 & 0xFFFFu) << 16),
                       (int)((td0 >> 16) | ((unsigned)(N & 0xFFFF) << 16)),
                       (int)(((unsigned)N >> 16) | (32u << 16)),
                       (int)(N & 0xFFFF),
                       (int)K, 0, 0 };
            v4i gz = { 0, 0, 0, 0 };
#if defined(__clang_major__) && __clang_major__ >= 23
            v8i gz8 = { 0, 0, 0, 0, 0, 0, 0, 0 };
            __builtin_amdgcn_tensor_load_to_lds(g0, g1, gz, gz, gz8, 0);
#else
            __builtin_amdgcn_tensor_load_to_lds(g0, g1, gz, gz, 0);
#endif
            __builtin_amdgcn_s_wait_tensorcnt(0);
        }
#else
        for (int idx = tid; idx < N * 4; idx += NT) {
            int n = idx >> 2, j = (idx & 3) * 8;
            const _Float16* sp = Wt + (size_t)n * K + kk + j;
            v8h d;
            if (kk + j + 7 < K) {
                d = *(const v8h*)sp;
            } else {
                #pragma unroll
                for (int q = 0; q < 8; ++q)
                    d[q] = (_Float16)((kk + j + q < K) ? (float)sp[q] : 0.f);
            }
            *(v8h*)&ldsB[n * 32 + j] = d;
        }
#endif
        __syncthreads();

        v8h alo = *(const v8h*)&ldsA[r * 32 + 16 * h];
        v8h ahi = *(const v8h*)&ldsA[r * 32 + 16 * h + 8];
        v16h a = __builtin_shufflevector(alo, ahi, 0,1,2,3,4,5,6,7,8,9,10,11,12,13,14,15);
        v16h bf = *(const v16h*)&ldsB[(16 * w + r) * 32 + 16 * h];
        acc = __builtin_amdgcn_wmma_f32_16x16x32_f16(false, a, false, bf,
                                                     (short)0, acc, false, false);
    }

    #pragma unroll
    for (int i = 0; i < 8; ++i) {
        int m = tileM + i + 8 * h, n = 16 * w + r;
        float vv = acc[i] + (BIAS ? bias[n] : 0.f);
        if (RELU) vv = fmaxf(vv, 0.f);
        C[(size_t)m * N + n] = vv;
    }
}

// ---------------------------------------------------------------------------
__global__ __launch_bounds__(1024)
void edge_norm_prep(const int* __restrict__ ei, float* __restrict__ norm,
                    float* __restrict__ dinv2) {
    __shared__ float deg[NN];
    __shared__ float dinv[NN];
    int t = threadIdx.x;
    if (t < NN) deg[t] = 1.f;
    __syncthreads();
    atomicAdd(&deg[ei[EE + t]], 1.f);
    __syncthreads();
    if (t < NN) { dinv[t] = rsqrtf(deg[t]); dinv2[t] = dinv[t] * dinv[t]; }
    __syncthreads();
    norm[t] = dinv[ei[t]] * dinv[ei[EE + t]];
}

__global__ void agg_init(const float* __restrict__ hW, const float* __restrict__ bias,
                         const float* __restrict__ dinv2, float* __restrict__ agg, int F) {
    size_t idx = (size_t)blockIdx.x * blockDim.x + threadIdx.x;
    size_t total = (size_t)BB * NN * F;
    if (idx >= total) return;
    size_t row = idx / F;
    int f = (int)(idx - row * F);
    agg[idx] = hW[idx] * dinv2[row % NN] + bias[f];
}

__global__ void agg_edges(const float* __restrict__ hW, const int* __restrict__ ei,
                          const float* __restrict__ norm, float* __restrict__ agg, int F) {
    int be = blockIdx.x;
    int b = be / EE, e = be - b * EE;
    int f = threadIdx.x;
    if (f >= F) return;
    int s = b * NN + ei[e];
    int d = b * NN + ei[EE + e];
    atomicAdd(&agg[(size_t)d * F + f], hW[(size_t)s * F + f] * norm[e]);
}

__global__ __launch_bounds__(32)
void ln_relu(const float* __restrict__ x, const float* __restrict__ gam,
             const float* __restrict__ bet, float* __restrict__ y, int F) {
    int row = blockIdx.x, lane = threadIdx.x;
    const float* xr = x + (size_t)row * F;
    float s = 0.f, s2 = 0.f;
    for (int f = lane; f < F; f += 32) { float v = xr[f]; s += v; s2 += v * v; }
    #pragma unroll
    for (int o = 16; o; o >>= 1) { s += __shfl_xor(s, o, 32); s2 += __shfl_xor(s2, o, 32); }
    float mean = s / F;
    float rstd = rsqrtf(s2 / F - mean * mean + 1e-5f);
    float* yr = y + (size_t)row * F;
    for (int f = lane; f < F; f += 32)
        yr[f] = fmaxf((xr[f] - mean) * rstd * gam[f] + bet[f], 0.f);
}

__global__ __launch_bounds__(32)
void ln_res_tanh(const float* __restrict__ x, const float* __restrict__ gam,
                 const float* __restrict__ bet, const float* __restrict__ iden,
                 float* __restrict__ xc, int coff) {
    const int F = OUTc;
    int row = blockIdx.x, lane = threadIdx.x;
    const float* xr = x + (size_t)row * F;
    float s = 0.f, s2 = 0.f;
    for (int f = lane; f < F; f += 32) { float v = xr[f]; s += v; s2 += v * v; }
    #pragma unroll
    for (int o = 16; o; o >>= 1) { s += __shfl_xor(s, o, 32); s2 += __shfl_xor(s2, o, 32); }
    float mean = s / F;
    float rstd = rsqrtf(s2 / F - mean * mean + 1e-5f);
    const float* ir = iden + (size_t)row * F;
    float* yr = xc + (size_t)row * (2 * OUTc) + coff;
    for (int f = lane; f < F; f += 32) {
        float v = (xr[f] - mean) * rstd * gam[f] + bet[f] + ir[f];
        yr[f] = tanhf(fmaxf(v, 0.f));
    }
}

__global__ __launch_bounds__(512)
void pool_kernel(const float* __restrict__ xc, float* __restrict__ xg) {
    int b = blockIdx.x, f = threadIdx.x;
    const float* base = xc + (size_t)b * NN * (2 * OUTc) + f;
    float s = 0.f, mx = -3.4e38f;
    for (int n = 0; n < NN; ++n) {
        float v = base[(size_t)n * (2 * OUTc)];
        s += v; mx = fmaxf(mx, v);
    }
    xg[(size_t)b * (4 * OUTc) + f] = s * (1.f / NN);
    xg[(size_t)b * (4 * OUTc) + 2 * OUTc + f] = mx;
}

__global__ void head_final(const float* __restrict__ h, const float* __restrict__ w2,
                           const float* __restrict__ b2, float* __restrict__ out) {
    int idx = blockIdx.x * blockDim.x + threadIdx.x;
    if (idx >= BB * 7) return;
    int b = idx / 7, j = idx - b * 7;
    const float* hr = h + (size_t)b * 96;
    float s = b2[j];
    for (int k = 0; k < 96; ++k) s += hr[k] * w2[k * 7 + j];
    out[idx] = s;
}

// ---------------------------------------------------------------------------
extern "C" void kernel_launch(void* const* d_in, const int* in_sizes, int n_in,
                              void* d_out, int out_size, void* d_ws, size_t ws_size,
                              hipStream_t stream) {
    (void)in_sizes; (void)n_in; (void)out_size; (void)ws_size;
    const float* data = (const float*)d_in[0];
    const int*   ei[2] = { (const int*)d_in[1], (const int*)d_in[2] };

    int ii = 3;
    const float *t1v1=(const float*)d_in[ii++], *t1g1=(const float*)d_in[ii++], *t1b1=(const float*)d_in[ii++];
    const float *t1v2=(const float*)d_in[ii++], *t1g2=(const float*)d_in[ii++], *t1b2=(const float*)d_in[ii++];
    const float *t1dw=(const float*)d_in[ii++], *t1db=(const float*)d_in[ii++];
    const float *t2v1=(const float*)d_in[ii++], *t2g1=(const float*)d_in[ii++], *t2b1=(const float*)d_in[ii++];
    const float *t2v2=(const float*)d_in[ii++], *t2g2=(const float*)d_in[ii++], *t2b2=(const float*)d_in[ii++];
    const float *t3v1=(const float*)d_in[ii++], *t3g1=(const float*)d_in[ii++], *t3b1=(const float*)d_in[ii++];
    const float *t3v2=(const float*)d_in[ii++], *t3g2=(const float*)d_in[ii++], *t3b2=(const float*)d_in[ii++];
    const float *t3dw=(const float*)d_in[ii++], *t3db=(const float*)d_in[ii++];
    const float *gw1[2], *gb1[2], *gw2[2], *gb2[2], *l1g[2], *l1b[2], *l2g[2], *l2b[2], *gpw[2], *gpb[2];
    for (int br = 0; br < 2; ++br) {
        gw1[br]=(const float*)d_in[ii++]; gb1[br]=(const float*)d_in[ii++];
        gw2[br]=(const float*)d_in[ii++]; gb2[br]=(const float*)d_in[ii++];
        l1g[br]=(const float*)d_in[ii++]; l1b[br]=(const float*)d_in[ii++];
        l2g[br]=(const float*)d_in[ii++]; l2b[br]=(const float*)d_in[ii++];
        gpw[br]=(const float*)d_in[ii++]; gpb[br]=(const float*)d_in[ii++];
    }
    const float *hw1=(const float*)d_in[ii++], *hb1=(const float*)d_in[ii++];
    const float *hw2=(const float*)d_in[ii++], *hb2=(const float*)d_in[ii++];

    // -------- workspace carve ------------------------------------------------
    char* p = (char*)d_ws;
    auto alloc = [&](size_t bytes) -> void* {
        void* r = (void*)p; p += (bytes + 255) & ~(size_t)255; return r;
    };
    const size_t S = (size_t)BB * 128 * TT * sizeof(float);
    float* bufA = (float*)alloc(S);
    float* bufB = (float*)alloc(S);
    float* bufC = (float*)alloc(S);
    float* bufR = (float*)alloc(S);
    const size_t Mn = (size_t)BB * NN;
    float* iden = (float*)alloc(Mn * OUTc * 4);
    float* hW   = (float*)alloc(Mn * OUTc * 4);
    float* agg  = (float*)alloc(Mn * OUTc * 4);
    float* h48  = (float*)alloc(Mn * HIDc * 4);
    float* xc   = (float*)alloc(Mn * 2 * OUTc * 4);
    float* xgr  = (float*)alloc((size_t)BB * 4 * OUTc * 4);
    float* hhd  = (float*)alloc((size_t)BB * 96 * 4);
    float* dinv2[2] = { (float*)alloc(NN * 4), (float*)alloc(NN * 4) };
    float* norm [2] = { (float*)alloc(EE * 4), (float*)alloc(EE * 4) };
    _Float16* w_t1c1 = (_Float16*)alloc((size_t)3*128*64*2);
    _Float16* w_t1c2 = (_Float16*)alloc((size_t)3*128*128*2);
    _Float16* w_t1ds = (_Float16*)alloc((size_t)128*64*2);
    _Float16* w_t2c1 = (_Float16*)alloc((size_t)3*128*128*2);
    _Float16* w_t2c2 = (_Float16*)alloc((size_t)3*128*128*2);
    _Float16* w_t3c1 = (_Float16*)alloc((size_t)3*64*128*2);
    _Float16* w_t3c2 = (_Float16*)alloc((size_t)3*64*64*2);
    _Float16* w_t3ds = (_Float16*)alloc((size_t)64*128*2);
    _Float16* wpwf[2] = { (_Float16*)alloc((size_t)TT*OUTc*2),  (_Float16*)alloc((size_t)TT*OUTc*2) };
    _Float16* ww1f[2] = { (_Float16*)alloc((size_t)TT*HIDc*2),  (_Float16*)alloc((size_t)TT*HIDc*2) };
    _Float16* ww2f[2] = { (_Float16*)alloc((size_t)HIDc*OUTc*2),(_Float16*)alloc((size_t)HIDc*OUTc*2) };
    _Float16* whf = (_Float16*)alloc((size_t)1024*96*2);

    // -------- weight prep ----------------------------------------------------
    wn_prep<<<2, 64, 0, stream>>>(t1v1, t1g1, w_t1c1, 128, 64, 3);
    wn_prep<<<2, 64, 0, stream>>>(t1v2, t1g2, w_t1c2, 128, 128, 3);
    cvt_f16<<<(128*64+255)/256, 256, 0, stream>>>(t1dw, w_t1ds, 128*64);
    wn_prep<<<2, 64, 0, stream>>>(t2v1, t2g1, w_t2c1, 128, 128, 3);
    wn_prep<<<2, 64, 0, stream>>>(t2v2, t2g2, w_t2c2, 128, 128, 3);
    wn_prep<<<1, 64, 0, stream>>>(t3v1, t3g1, w_t3c1, 64, 128, 3);
    wn_prep<<<1, 64, 0, stream>>>(t3v2, t3g2, w_t3c2, 64, 64, 3);
    cvt_f16<<<(64*128+255)/256, 256, 0, stream>>>(t3dw, w_t3ds, 64*128);
    for (int br = 0; br < 2; ++br) {   // GEMM weights pre-TRANSPOSED: [N][K]
        cvt_f16_t<<<(TT*OUTc+255)/256, 256, 0, stream>>>(gpw[br], wpwf[br], TT, OUTc);
        cvt_f16_t<<<(TT*HIDc+255)/256, 256, 0, stream>>>(gw1[br], ww1f[br], TT, HIDc);
        cvt_f16_t<<<(HIDc*OUTc+255)/256, 256, 0, stream>>>(gw2[br], ww2f[br], HIDc, OUTc);
    }
    cvt_f16_t<<<(1024*96+255)/256, 256, 0, stream>>>(hw1, whf, 1024, 96);

    // -------- TCN ------------------------------------------------------------
    const dim3 cg(TT/16, BB);
    conv_wmma_t<64,128,3,1,1,0><<<cg, 256, 0, stream>>>(data, w_t1c1, t1b1, data, bufC);
    conv_wmma_t<64,128,1,1,0,0><<<cg, 256, 0, stream>>>(data, w_t1ds, t1db, data, bufR);
    conv_wmma_t<128,128,3,1,1,1><<<cg, 256, 0, stream>>>(bufC, w_t1c2, t1b2, bufR, bufA);
    conv_wmma_t<128,128,3,2,1,0><<<cg, 256, 0, stream>>>(bufA, w_t2c1, t2b1, bufA, bufC);
    conv_wmma_t<128,128,3,2,1,1><<<cg, 256, 0, stream>>>(bufC, w_t2c2, t2b2, bufA, bufB);
    conv_wmma_t<128,64,3,4,1,0><<<cg, 128, 0, stream>>>(bufB, w_t3c1, t3b1, bufB, bufC);
    conv_wmma_t<128,64,1,1,0,0><<<cg, 128, 0, stream>>>(bufB, w_t3ds, t3db, bufB, bufR);
    conv_wmma_t<64,64,3,4,1,1><<<cg, 128, 0, stream>>>(bufC, w_t3c2, t3b2, bufR, bufA);
    const float* xg = bufA;                       // [B*N, T] contiguous

    // -------- GCN branches ---------------------------------------------------
    for (int br = 0; br < 2; ++br) {
        edge_norm_prep<<<1, 1024, 0, stream>>>(ei[br], norm[br], dinv2[br]);
        gemm_wmma_t<512,256,0,1><<<Mn/16, 512, 0, stream>>>(xg, wpwf[br], gpb[br], iden);
        gemm_wmma_t<512,48,0,0><<<Mn/16, 96, 0, stream>>>(xg, ww1f[br], nullptr, hW);
        agg_init<<<(int)((Mn*HIDc + 255)/256), 256, 0, stream>>>(hW, gb1[br], dinv2[br], agg, HIDc);
        agg_edges<<<BB*EE, 64, 0, stream>>>(hW, ei[br], norm[br], agg, HIDc);
        ln_relu<<<(int)Mn, 32, 0, stream>>>(agg, l1g[br], l1b[br], h48, HIDc);
        gemm_wmma_t<48,256,0,0><<<Mn/16, 512, 0, stream>>>(h48, ww2f[br], nullptr, hW);
        agg_init<<<(int)((Mn*OUTc + 255)/256), 256, 0, stream>>>(hW, gb2[br], dinv2[br], agg, OUTc);
        agg_edges<<<BB*EE, 256, 0, stream>>>(hW, ei[br], norm[br], agg, OUTc);
        ln_res_tanh<<<(int)Mn, 32, 0, stream>>>(agg, l2g[br], l2b[br], iden, xc, br * OUTc);
    }

    // -------- pooling + head -------------------------------------------------
    pool_kernel<<<BB, 512, 0, stream>>>(xc, xgr);
    gemm_wmma_t<1024,96,1,1><<<BB/16, 192, 0, stream>>>(xgr, whf, hb1, hhd);
    head_final<<<(BB*7 + 63)/64, 64, 0, stream>>>(hhd, hw2, hb2, (float*)d_out);
}
